// SPIRED_Fitness_Union_81209241632970
// MI455X (gfx1250) — matrix operand order repaired and checked by reading, not compiled, not run on hardware
//
#include <hip/hip_runtime.h>
#include <hip/hip_bf16.h>

// ---------------------------------------------------------------------------
// SPIRED-Fitness union block on MI455X (gfx1250, wave32).
// All GEMM-like ops: v_wmma_f32_16x16x32_bf16 (f32 accumulate).
// Big 136->128 conv3x3: implicit GEMM; input window staged to LDS with
// global_load_async_to_lds_b128 (ASYNCcnt), fragment-major bf16 weights,
// 4 M-tiles per wave for B reuse, b128/ds_b128 fragment loads throughout.
// ---------------------------------------------------------------------------

typedef __attribute__((ext_vector_type(16))) __bf16 v16bf;
typedef __attribute__((ext_vector_type(8)))  float  v8f;

constexpr int LI   = 384;
constexpr int PIXI = LI * LI;        // 147456
constexpr int NODEI= 256;
constexpr int HI   = 8;
constexpr int HDI  = 32;
constexpr int PDI  = 128;
constexpr int D1I  = 1280;
constexpr int CPI  = 160;            // conv channels padded 136 -> 160
constexpr int LPI  = LI + 2;         // 386 (spatial zero pad)
constexpr int CONV_W = 66;           // staged pixels per row (64 out + 2 halo)

__device__ __forceinline__ float leakyf(float v) { return v >= 0.f ? v : 0.01f * v; }

__device__ __forceinline__ v8f wmma_bf16(v16bf a, v16bf b, v8f c) {
  return __builtin_amdgcn_wmma_f32_16x16x32_bf16(false, a, false, b, (short)0, c,
                                                 false, false);
}

union frag_u { uint4 u[2]; v16bf v; };

// A fragment: row-major 16x32 tile (M x K), lda in elements (16B-aligned rows).
// lane m holds row m; K 0-15 (lanes 0-15) or 16-31 (lanes 16-31).
// Two contiguous 16B loads per lane (global_load_b128 or ds_load_b128).
__device__ __forceinline__ v16bf load_A(const __bf16* A, int lda) {
  int lane = threadIdx.x & 31;
  int m = lane & 15;
  int kb = (lane & 16) ? 16 : 0;
  const __bf16* row = A + (size_t)m * lda + kb;
  frag_u f;
  f.u[0] = *(const uint4*)(row);
  f.u[1] = *(const uint4*)(row + 8);
  return f.v;
}

// B fragment from fragment-major storage: lane's 16 values contiguous (32B).
__device__ __forceinline__ v16bf load_Bfrag(const __bf16* Bf) {
  int lane = threadIdx.x & 31;
  const __bf16* p = Bf + lane * 16;
  frag_u f;
  f.u[0] = *(const uint4*)(p);
  f.u[1] = *(const uint4*)(p + 8);
  return f.v;
}

// B fragment from an N-major (transposed) source: row n = column n of B,
// element j = Bt[n*ldt + kb2 + j]; contiguous 32B per lane.
__device__ __forceinline__ v16bf load_BT(const __bf16* Bt, int ldt) {
  int lane = threadIdx.x & 31;
  int n = lane & 15;
  int kb2 = (lane & 16) ? 16 : 0;
  const __bf16* p = Bt + (size_t)n * ldt + kb2;
  frag_u f;
  f.u[0] = *(const uint4*)(p);
  f.u[1] = *(const uint4*)(p + 8);
  return f.v;
}

// ---------------------------------------------------------------------------
__global__ void k_f32_to_bf16_v4(const float* __restrict__ src, __bf16* __restrict__ dst,
                                 int n4) {
  int i = blockIdx.x * blockDim.x + threadIdx.x;
  if (i >= n4) return;
  float4 fv = ((const float4*)src)[i];
  union { uint2 u; __bf16 h[4]; } u;
  u.h[0] = (__bf16)fv.x; u.h[1] = (__bf16)fv.y; u.h[2] = (__bf16)fv.z; u.h[3] = (__bf16)fv.w;
  ((uint2*)dst)[i] = u.u;
}

// Repack f32 row-major (K,N) weights into bf16 fragment-major 32x16 tiles.
__global__ void k_pack_w(const float* __restrict__ src, __bf16* __restrict__ dst,
                         int N, int K) {
  int i = blockIdx.x * blockDim.x + threadIdx.x;
  if (i >= K * N) return;
  int j    = i & 15;
  int lane = (i >> 4) & 31;
  int tile = i >> 9;
  int ntiles = N >> 4;
  int kt = tile / ntiles, nt = tile - kt * ntiles;
  int kk = kt * 32 + ((lane & 16) ? 16 : 0) + j;
  int n  = nt * 16 + (lane & 15);
  dst[i] = (__bf16)src[(size_t)kk * N + n];
}

// Generic GEMM, fragment-major B: C[M,N] = A@B (+bias). Grid (M/16, N/16), block 32.
__global__ void k_gemm(const __bf16* __restrict__ A, const __bf16* __restrict__ Bf,
                       const float* __restrict__ bias, float* __restrict__ C,
                       int N, int K) {
  int mt = blockIdx.x, nt = blockIdx.y;
  int ntiles = N >> 4;
  const __bf16* Ab = A + (size_t)mt * 16 * K;
  v8f acc = {};
  for (int kt = 0; kt < (K >> 5); ++kt)
    acc = wmma_bf16(load_A(Ab + kt * 32, K),
                    load_Bfrag(Bf + ((size_t)(kt * ntiles + nt) << 9)), acc);
  int lane = threadIdx.x & 31, n = lane & 15, mo = (lane & 16) ? 8 : 0;
  float bv = bias ? bias[nt * 16 + n] : 0.f;
#pragma unroll
  for (int r = 0; r < 8; ++r)
    C[(size_t)(mt * 16 + mo + r) * N + nt * 16 + n] = acc[r] + bv;
}

// Attention scores: aout[8+h][l][m] = q_h(l,:)·k_h(m,:) / sqrt(32). Grid (24,24,8).
__global__ void k_scores(const __bf16* __restrict__ qb, const __bf16* __restrict__ kb,
                         float* __restrict__ aout) {
  int lt = blockIdx.x, mt = blockIdx.y, h = blockIdx.z;
  v16bf a = load_A(qb + (size_t)(lt * 16) * NODEI + h * HDI, NODEI);
  v16bf b = load_BT(kb + (size_t)(mt * 16) * NODEI + h * HDI, NODEI);
  v8f acc = {};
  acc = wmma_bf16(a, b, acc);
  int lane = threadIdx.x & 31, n = lane & 15, mo = (lane & 16) ? 8 : 0;
  float* ob = aout + (size_t)(8 + h) * PIXI + (size_t)(lt * 16) * LI + mt * 16;
#pragma unroll
  for (int r = 0; r < 8; ++r) ob[(size_t)(mo + r) * LI + n] = acc[r] * 0.17677669529663687f;
}

// a_pair: aout[h][pix] = z[pix,:] @ Wp2a[:,h]. Grid 576, block 256.
__global__ void k_apair(const float* __restrict__ z, const float* __restrict__ Wp2a,
                        float* __restrict__ aout) {
  __shared__ float wsh[PDI * HI];
  for (int i = threadIdx.x; i < PDI * HI; i += blockDim.x) wsh[i] = Wp2a[i];
  __syncthreads();
  int pix = blockIdx.x * blockDim.x + threadIdx.x;
  const float* zr = z + (size_t)pix * PDI;
  float acc[HI] = {};
  for (int p = 0; p < PDI; ++p) {
    float zv = zr[p];
#pragma unroll
    for (int h = 0; h < HI; ++h) acc[h] += zv * wsh[p * HI + h];
  }
#pragma unroll
  for (int h = 0; h < HI; ++h) aout[(size_t)h * PIXI + pix] = acc[h];
}

// Instance-norm stats, planar (C,384,384): one block per channel.
__global__ void k_stats_planar(const float* __restrict__ src, float* __restrict__ out) {
  int c = blockIdx.x;
  const float* p = src + (size_t)c * PIXI;
  float s = 0.f, s2 = 0.f;
  for (int i = threadIdx.x; i < PIXI; i += blockDim.x) { float v = p[i]; s += v; s2 += v * v; }
  __shared__ float r1[256], r2[256];
  r1[threadIdx.x] = s; r2[threadIdx.x] = s2; __syncthreads();
  for (int st = 128; st > 0; st >>= 1) {
    if (threadIdx.x < st) { r1[threadIdx.x] += r1[threadIdx.x + st]; r2[threadIdx.x] += r2[threadIdx.x + st]; }
    __syncthreads();
  }
  if (threadIdx.x == 0) {
    float m = r1[0] / (float)PIXI;
    float var = r2[0] / (float)PIXI - m * m;
    out[2 * c] = m; out[2 * c + 1] = rsqrtf(var + 1e-5f);
  }
}

// Instance-norm stats, NHWC with C interleaved channels.
__global__ void k_stats_nhwc(const float* __restrict__ src, int C, float* __restrict__ out) {
  int c = blockIdx.x;
  float s = 0.f, s2 = 0.f;
  for (int i = threadIdx.x; i < PIXI; i += blockDim.x) {
    float v = src[(size_t)i * C + c]; s += v; s2 += v * v;
  }
  __shared__ float r1[256], r2[256];
  r1[threadIdx.x] = s; r2[threadIdx.x] = s2; __syncthreads();
  for (int st = 128; st > 0; st >>= 1) {
    if (threadIdx.x < st) { r1[threadIdx.x] += r1[threadIdx.x + st]; r2[threadIdx.x] += r2[threadIdx.x + st]; }
    __syncthreads();
  }
  if (threadIdx.x == 0) {
    float m = r1[0] / (float)PIXI;
    float var = r2[0] / (float)PIXI - m * m;
    out[2 * c] = m; out[2 * c + 1] = rsqrtf(var + 1e-5f);
  }
}

// Small direct conv3x3 (Cin<=16 -> 8) with fused instance-norm + leaky.
// Optional secondary bf16 per-head planar output (for WMMA A fragments).
__global__ void k_conv_small(const float* __restrict__ in, const float* __restrict__ stats,
                             const float* __restrict__ W, const float* __restrict__ bias,
                             float* __restrict__ out, __bf16* __restrict__ out_bp,
                             int Cin, int out_nhwc) {
  __shared__ float wl[8 * 16 * 9];
  __shared__ float bl[8];
  int tot = 8 * Cin * 9;
  for (int i = threadIdx.x; i < tot; i += blockDim.x) wl[i] = W[i];
  if (threadIdx.x < 8) bl[threadIdx.x] = bias[threadIdx.x];
  __syncthreads();
  int pix = blockIdx.x * blockDim.x + threadIdx.x;
  int y = pix / LI, x = pix % LI;
  float acc[8];
#pragma unroll
  for (int co = 0; co < 8; ++co) acc[co] = bl[co];
  for (int ci = 0; ci < Cin; ++ci) {
    float m = stats[2 * ci], inv = stats[2 * ci + 1];
    const float* pl = in + (size_t)ci * PIXI;
    float v[9];
    int t = 0;
    for (int dy = -1; dy <= 1; ++dy)
      for (int dx = -1; dx <= 1; ++dx) {
        int yy = y + dy, xx = x + dx;
        v[t++] = (yy >= 0 && yy < LI && xx >= 0 && xx < LI) ? (pl[yy * LI + xx] - m) * inv : 0.f;
      }
#pragma unroll
    for (int co = 0; co < 8; ++co) {
      const float* wr = wl + (co * Cin + ci) * 9;
#pragma unroll
      for (int tt = 0; tt < 9; ++tt) acc[co] += v[tt] * wr[tt];
    }
  }
#pragma unroll
  for (int co = 0; co < 8; ++co) {
    float o = leakyf(acc[co]);
    if (out_nhwc) {
      out[(size_t)pix * 8 + co] = o;
      if (out_bp) out_bp[(size_t)co * PIXI + pix] = (__bf16)o;
    } else {
      out[(size_t)co * PIXI + pix] = o;
    }
  }
}

// Transpose v (384x256 f32) -> vT (256x384 bf16) for contiguous B fragments.
__global__ void k_transpose_vT(const float* __restrict__ vf, __bf16* __restrict__ vT) {
  int i = blockIdx.x * blockDim.x + threadIdx.x;
  if (i >= NODEI * LI) return;
  int d = i / LI, m = i - d * LI;
  vT[i] = (__bf16)vf[(size_t)m * NODEI + d];
}

// nfn: cat[l, 1024 + h*32 + d] = sum_m a[l,m,h] * v[m, h*32+d]. Grid (24,2,8).
__global__ void k_nfn(const __bf16* __restrict__ apl, const __bf16* __restrict__ vT,
                      float* __restrict__ cat) {
  int lt = blockIdx.x, dt = blockIdx.y, h = blockIdx.z;
  const __bf16* Ab = apl + (size_t)h * PIXI + (size_t)(lt * 16) * LI;
  const __bf16* Bb = vT + (size_t)(h * HDI + dt * 16) * LI;
  v8f acc = {};
  for (int k0 = 0; k0 < LI; k0 += 32)
    acc = wmma_bf16(load_A(Ab + k0, LI), load_BT(Bb + k0, LI), acc);
  int lane = threadIdx.x & 31, n = lane & 15, mo = (lane & 16) ? 8 : 0;
  float* ob = cat + (size_t)(lt * 16) * D1I + 1024 + h * HDI + dt * 16;
#pragma unroll
  for (int r = 0; r < 8; ++r) ob[(size_t)(mo + r) * D1I + n] = acc[r];
}

// nfp: cat[l, h*128 + p] = sum_m a[l,m,h] * z[l,m,p]. Grid (384,8).
__global__ void k_nfp(const __bf16* __restrict__ apl, const __bf16* __restrict__ zb,
                      float* __restrict__ cat) {
  int l = blockIdx.x, pt = blockIdx.y;
  int lane = threadIdx.x & 31, n = lane & 15;
  int kb2 = (lane & 16) ? 16 : 0;
  v8f acc = {};
  for (int k0 = 0; k0 < LI; k0 += 32) {
    v16bf a = load_A(apl + (size_t)l * LI + k0, PIXI);  // lda = plane stride
    v16bf b;
    const __bf16* bb = zb + ((size_t)l * LI + k0 + kb2) * PDI + pt * 16 + n;
#pragma unroll
    for (int j = 0; j < 16; ++j) b[j] = bb[(size_t)j * PDI];
    acc = wmma_bf16(a, b, acc);
  }
  int mo = (lane & 16) ? 8 : 0;
#pragma unroll
  for (int r = 0; r < 8; ++r) {
    int hh = mo + r;
    if (hh < 8) cat[(size_t)l * D1I + hh * PDI + pt * 16 + n] = acc[r];
  }
}

// Row LayerNorm -> bf16 (optional leaky). One block per row.
__global__ void k_ln_bf16(const float* __restrict__ in, const float* __restrict__ w,
                          const float* __restrict__ b, __bf16* __restrict__ out,
                          int C, int act) {
  int row = blockIdx.x;
  const float* p = in + (size_t)row * C;
  float s = 0.f, s2 = 0.f;
  for (int i = threadIdx.x; i < C; i += blockDim.x) { float v = p[i]; s += v; s2 += v * v; }
  __shared__ float r1[256], r2[256];
  __shared__ float mi[2];
  r1[threadIdx.x] = s; r2[threadIdx.x] = s2; __syncthreads();
  for (int st = 128; st > 0; st >>= 1) {
    if (threadIdx.x < st) { r1[threadIdx.x] += r1[threadIdx.x + st]; r2[threadIdx.x] += r2[threadIdx.x + st]; }
    __syncthreads();
  }
  if (threadIdx.x == 0) {
    float m = r1[0] / (float)C;
    float var = r2[0] / (float)C - m * m;
    mi[0] = m; mi[1] = rsqrtf(var + 1e-5f);
  }
  __syncthreads();
  for (int i = threadIdx.x; i < C; i += blockDim.x) {
    float v = (p[i] - mi[0]) * mi[1] * w[i] + b[i];
    if (act) v = leakyf(v);
    out[(size_t)row * C + i] = (__bf16)v;
  }
}

// Final: x_new = LN(x + h2). C = 256, block 256, grid 384.
__global__ void k_ln_final(const float* __restrict__ x, const float* __restrict__ h2,
                           const float* __restrict__ w, const float* __restrict__ b,
                           float* __restrict__ out) {
  int row = blockIdx.x, i = threadIdx.x;
  float v = x[(size_t)row * 256 + i] + h2[(size_t)row * 256 + i];
  __shared__ float r1[256], r2[256];
  r1[i] = v; r2[i] = v * v; __syncthreads();
  for (int st = 128; st > 0; st >>= 1) {
    if (i < st) { r1[i] += r1[i + st]; r2[i] += r2[i + st]; }
    __syncthreads();
  }
  float m = r1[0] / 256.f;
  float var = r2[0] / 256.f - m * m;
  out[(size_t)row * 256 + i] = (v - m) * rsqrtf(var + 1e-5f) * w[i] + b[i];
}

// Build padded, instance-normed bf16 NHWC conv input (386,386,160).
// One thread = one 16-channel chunk -> two b128 stores.
__global__ void k_pack_zp(const float* __restrict__ z, const float* __restrict__ afin,
                          const float* __restrict__ statp, __bf16* __restrict__ zp) {
  int t = blockIdx.x * blockDim.x + threadIdx.x;
  if (t >= LPI * LPI * 10) return;
  int chunk = t % 10;
  int pix = t / 10;
  int y = pix / LPI, x = pix % LPI;
  bool inside = (y >= 1 && y <= LI && x >= 1 && x <= LI);
  int sp = (y - 1) * LI + (x - 1);
  int c0 = chunk * 16;
  union { uint4 q[2]; __bf16 h[16]; } o;
#pragma unroll
  for (int j = 0; j < 16; ++j) {
    int c = c0 + j;
    float v = 0.f;
    if (inside && c < PDI + HI) {
      float raw = (c < PDI) ? z[(size_t)sp * PDI + c] : afin[(size_t)sp * HI + (c - PDI)];
      v = (raw - statp[2 * c]) * statp[2 * c + 1];
    }
    o.h[j] = (__bf16)v;
  }
  uint4* d = (uint4*)(zp + (size_t)pix * CPI + c0);
  d[0] = o.q[0];
  d[1] = o.q[1];
}

// Repack conv_p weights OIHW (128,136,3,3) -> bf16 fragment-major:
// tile = (tap*5 + kcb)*8 + ot ; elem = lane*16 + j.
__global__ void k_pack_wp(const float* __restrict__ wsrc, __bf16* __restrict__ wdst) {
  int i = blockIdx.x * blockDim.x + threadIdx.x;
  if (i >= 9 * 5 * 8 * 512) return;
  int j    = i & 15;
  int lane = (i >> 4) & 31;
  int tile = i >> 9;
  int ot  = tile & 7;
  int tk  = tile >> 3;
  int kcb = tk % 5;
  int tap = tk / 5;
  int c = kcb * 32 + ((lane & 16) ? 16 : 0) + j;
  int o = ot * 16 + (lane & 15);
  float v = (c < 136) ? wsrc[((size_t)o * 136 + c) * 9 + tap] : 0.f;
  wdst[i] = (__bf16)v;
}

// Implicit-GEMM conv3x3 136->128 + bias + leaky.
// Grid (6,384), block 256 (8 waves = 8 o-tiles), dynamic LDS = 3*66*160 bf16.
// Stage: async-copy the 3-row input window to LDS once (ASYNCcnt), barrier.
// Compute: per wave 4 M-tiles x 16 out-ch; A fragments from LDS (ds_load_b128),
// each weight fragment loaded once and feeds 4 WMMAs (180 WMMA/wave).
__global__ void k_conv_big(const __bf16* __restrict__ zp, const __bf16* __restrict__ wPf,
                           const float* __restrict__ bias, float* __restrict__ out) {
  extern __shared__ __bf16 lds[];            // [3][CONV_W][CPI]
  int xg = blockIdx.x;                       // x-group: pixels xg*64 .. xg*64+63
  int y  = blockIdx.y;
  int xbase = xg * 64;

  // Cooperative stage: rows y..y+2, 66 pixels each, contiguous 160ch runs.
  constexpr int ROWE = CONV_W * CPI;         // 10560 elements per row
  constexpr int NELEM = 3 * ROWE;            // 31680 elements (63360 B)
  for (int e = threadIdx.x * 8; e < NELEM; e += blockDim.x * 8) {
    int row = e / ROWE;
    int rem = e - row * ROWE;
    const __bf16* g = zp + ((size_t)(y + row) * LPI + xbase) * CPI + rem;
#if defined(__gfx1250__)
    unsigned int loff = (unsigned int)(size_t)(lds + e);
    asm volatile("global_load_async_to_lds_b128 %0, %1, off"
                 :: "v"(loff), "v"(g) : "memory");
#else
    *(uint4*)(lds + e) = *(const uint4*)g;   // host-parse fallback
#endif
  }
#if defined(__gfx1250__)
  asm volatile("s_wait_asynccnt 0x0" ::: "memory");
#endif
  __syncthreads();

  int wv = threadIdx.x >> 5;                 // o-tile
  int o0 = wv * 16;
  v8f acc0 = {}, acc1 = {}, acc2 = {}, acc3 = {};
  for (int tap = 0; tap < 9; ++tap) {
    int dy = tap / 3, dx = tap - dy * 3;
    const __bf16* arow = lds + ((size_t)dy * CONV_W + dx) * CPI;
    const __bf16* btap = wPf + (((size_t)tap * 5) * 8 + wv) * 512;
    for (int kcb = 0; kcb < 5; ++kcb) {
      v16bf b = load_Bfrag(btap + (size_t)kcb * 8 * 512);
      int kc = kcb * 32;
      acc0 = wmma_bf16(load_A(arow + (size_t)0  * CPI + kc, CPI), b, acc0);
      acc1 = wmma_bf16(load_A(arow + (size_t)16 * CPI + kc, CPI), b, acc1);
      acc2 = wmma_bf16(load_A(arow + (size_t)32 * CPI + kc, CPI), b, acc2);
      acc3 = wmma_bf16(load_A(arow + (size_t)48 * CPI + kc, CPI), b, acc3);
    }
  }
  int lane = threadIdx.x & 31, n = lane & 15, mo = (lane & 16) ? 8 : 0;
  float bv = bias[o0 + n];
  v8f accs[4] = {acc0, acc1, acc2, acc3};
#pragma unroll
  for (int s = 0; s < 4; ++s) {
#pragma unroll
    for (int r = 0; r < 8; ++r) {
      int x = xbase + s * 16 + mo + r;
      out[((size_t)y * LI + x) * PDI + o0 + n] = leakyf(accs[s][r] + bv);
    }
  }
}

// ---------------------------------------------------------------------------
extern "C" void kernel_launch(void* const* d_in, const int* in_sizes, int n_in,
                              void* d_out, int out_size, void* d_ws, size_t ws_size,
                              hipStream_t stream) {
  const float* x     = (const float*)d_in[0];
  const float* z     = (const float*)d_in[1];
  const float* plddt = (const float*)d_in[2];
  const float* Wq    = (const float*)d_in[3];
  const float* Wk    = (const float*)d_in[4];
  const float* Wv    = (const float*)d_in[5];
  const float* Wp2a  = (const float*)d_in[6];
  const float* caw   = (const float*)d_in[7];
  const float* cab   = (const float*)d_in[8];
  const float* cmw   = (const float*)d_in[9];
  const float* cmb   = (const float*)d_in[10];
  const float* ln1w  = (const float*)d_in[11];
  const float* ln1b  = (const float*)d_in[12];
  const float* l1w   = (const float*)d_in[13];
  const float* l1b   = (const float*)d_in[14];
  const float* ln2w  = (const float*)d_in[15];
  const float* ln2b  = (const float*)d_in[16];
  const float* l2w   = (const float*)d_in[17];
  const float* l2b   = (const float*)d_in[18];
  const float* lnfw  = (const float*)d_in[19];
  const float* lnfb  = (const float*)d_in[20];
  const float* cpw   = (const float*)d_in[21];
  const float* cpb   = (const float*)d_in[22];

  float* out   = (float*)d_out;
  float* x_new = out;
  float* z_new = out + 98304;

  char* base = (char*)d_ws;
  size_t off = 0;
  auto alloc = [&](size_t bytes) -> void* {
    off = (off + 255) & ~(size_t)255;
    void* p = base + off;
    off += bytes;
    return p;
  };

  __bf16* xb    = (__bf16*)alloc((size_t)384 * 256 * 2);
  __bf16* Wqf   = (__bf16*)alloc((size_t)256 * 256 * 2);   // fragment-major
  __bf16* Wkf   = (__bf16*)alloc((size_t)256 * 256 * 2);
  __bf16* Wvf   = (__bf16*)alloc((size_t)256 * 256 * 2);
  __bf16* l1wf  = (__bf16*)alloc((size_t)1280 * 512 * 2);
  __bf16* l2wf  = (__bf16*)alloc((size_t)512 * 256 * 2);
  __bf16* zb    = (__bf16*)alloc((size_t)PIXI * PDI * 2);
  float*  qf    = (float*) alloc((size_t)384 * 256 * 4);
  float*  kf    = (float*) alloc((size_t)384 * 256 * 4);
  float*  vf    = (float*) alloc((size_t)384 * 256 * 4);
  __bf16* qh    = (__bf16*)alloc((size_t)384 * 256 * 2);
  __bf16* kh    = (__bf16*)alloc((size_t)384 * 256 * 2);
  __bf16* vT    = (__bf16*)alloc((size_t)256 * 384 * 2);   // transposed bf16 v
  float*  aout  = (float*) alloc((size_t)16 * PIXI * 4);
  float*  stat_a= (float*) alloc(32 * 4);
  float*  cm_in = (float*) alloc((size_t)9 * PIXI * 4);
  float*  stat_m= (float*) alloc(18 * 4);
  float*  afin  = (float*) alloc((size_t)PIXI * HI * 4);
  __bf16* apl   = (__bf16*)alloc((size_t)16 * PIXI * 2);   // per-head planar (8 + 8 zero)
  float*  cat   = (float*) alloc((size_t)384 * D1I * 4);
  __bf16* h0b   = (__bf16*)alloc((size_t)384 * D1I * 2);
  float*  h1    = (float*) alloc((size_t)384 * 512 * 4);
  __bf16* h1b   = (__bf16*)alloc((size_t)384 * 512 * 2);
  float*  h2    = (float*) alloc((size_t)384 * 256 * 4);
  float*  statp = (float*) alloc(272 * 4);
  __bf16* zp    = (__bf16*)alloc((size_t)LPI * LPI * CPI * 2);
  __bf16* wPf   = (__bf16*)alloc((size_t)9 * 5 * 8 * 512 * 2);
  (void)ws_size; (void)in_sizes; (void)n_in; (void)out_size;

  auto cvt4 = [&](const float* s, __bf16* d, int n) {
    int n4 = n >> 2;
    k_f32_to_bf16_v4<<<(n4 + 255) / 256, 256, 0, stream>>>(s, d, n4);
  };
  auto packw = [&](const float* s, __bf16* d, int N, int K) {
    k_pack_w<<<(N * K + 255) / 256, 256, 0, stream>>>(s, d, N, K);
  };

  // ---- bf16 staging: activations row-major, weights fragment-major
  cvt4(x, xb, 384 * 256);
  packw(Wq, Wqf, 256, 256);
  packw(Wk, Wkf, 256, 256);
  packw(Wv, Wvf, 256, 256);
  packw(l1w, l1wf, 512, 1280);
  packw(l2w, l2wf, 256, 512);
  cvt4(z, zb, PIXI * PDI);
  hipMemsetAsync(apl + (size_t)8 * PIXI, 0, (size_t)8 * PIXI * 2, stream);

  // ---- q, k, v projections (WMMA)
  k_gemm<<<dim3(24, 16), 32, 0, stream>>>(xb, Wqf, nullptr, qf, 256, 256);
  k_gemm<<<dim3(24, 16), 32, 0, stream>>>(xb, Wkf, nullptr, kf, 256, 256);
  k_gemm<<<dim3(24, 16), 32, 0, stream>>>(xb, Wvf, nullptr, vf, 256, 256);
  cvt4(qf, qh, 384 * 256);
  cvt4(kf, kh, 384 * 256);
  k_transpose_vT<<<(256 * 384 + 255) / 256, 256, 0, stream>>>(vf, vT);

  // ---- attention map inputs: channels 0-7 pair, 8-15 node
  k_apair<<<PIXI / 256, 256, 0, stream>>>(z, Wp2a, aout);
  k_scores<<<dim3(24, 24, 8), 32, 0, stream>>>(qh, kh, aout);

  // ---- conv_a (16 -> 8) with fused instance norm + leaky -> planar cm_in
  k_stats_planar<<<16, 256, 0, stream>>>(aout, stat_a);
  k_conv_small<<<PIXI / 256, 256, 0, stream>>>(aout, stat_a, caw, cab, cm_in, nullptr, 16, 0);

  // ---- append plddt as channel 8, then conv_m (9 -> 8) -> afin (NHWC) + apl (planar bf16)
  hipMemcpyAsync(cm_in + (size_t)8 * PIXI, plddt, (size_t)PIXI * 4,
                 hipMemcpyDeviceToDevice, stream);
  k_stats_planar<<<9, 256, 0, stream>>>(cm_in, stat_m);
  k_conv_small<<<PIXI / 256, 256, 0, stream>>>(cm_in, stat_m, cmw, cmb, afin, apl, 9, 1);

  // ---- aggregations into cat = [nfp(1024) | nfn(256)] (WMMA)
  k_nfp<<<dim3(384, 8), 32, 0, stream>>>(apl, zb, cat);
  k_nfn<<<dim3(24, 2, 8), 32, 0, stream>>>(apl, vT, cat);

  // ---- node MLP: LN -> lin1 -> LN+leaky -> lin2 -> residual LN
  k_ln_bf16<<<384, 256, 0, stream>>>(cat, ln1w, ln1b, h0b, 1280, 0);
  k_gemm<<<dim3(24, 32), 32, 0, stream>>>(h0b, l1wf, l1b, h1, 512, 1280);
  k_ln_bf16<<<384, 256, 0, stream>>>(h1, ln2w, ln2b, h1b, 512, 1);
  k_gemm<<<dim3(24, 16), 32, 0, stream>>>(h1b, l2wf, l2b, h2, 256, 512);
  k_ln_final<<<384, 256, 0, stream>>>(x, h2, lnfw, lnfb, x_new);

  // ---- pair update: stats, packed+normed bf16 NHWC input, LDS-staged implicit-GEMM conv
  k_stats_nhwc<<<128, 256, 0, stream>>>(z, PDI, statp);
  k_stats_nhwc<<<8, 256, 0, stream>>>(afin, HI, statp + 256);
  k_pack_zp<<<(LPI * LPI * 10 + 255) / 256, 256, 0, stream>>>(z, afin, statp, zp);
  k_pack_wp<<<(9 * 5 * 8 * 512 + 255) / 256, 256, 0, stream>>>(cpw, wPf);
  k_conv_big<<<dim3(6, 384), 256, (size_t)3 * CONV_W * CPI * 2, stream>>>(zp, wPf, cpb, z_new);
}
